// Inference_82025285419171
// MI455X (gfx1250) — matrix operand order repaired and checked by reading, not compiled, not run on hardware
//
#include <hip/hip_runtime.h>
#include <hip/hip_bf16.h>
#include <stdint.h>
#include <stddef.h>

// ---------------------------------------------------------------------------
// Problem constants (from reference: b=4, n=96, D=768, H=4, DH=192)
// ---------------------------------------------------------------------------
#define D_    768
#define D2_   1536
#define NB_   4
#define NS_   96
#define NH_   4
#define DH_   192
#define NTOK  (NB_*NS_*NS_)          // 36864 rows
#define NEGV  (-10000.0f)

typedef __attribute__((ext_vector_type(16))) __bf16 bf16x16;
typedef __attribute__((ext_vector_type(8)))  float  f32x8;
typedef __attribute__((ext_vector_type(4)))  float  f32x4;

union FragAB { bf16x16 v; f32x4 q[2]; };
union V4U8   { f32x4 v; unsigned short u[8]; };

__device__ __forceinline__ unsigned short f2bf(float x) {
  unsigned u = __float_as_uint(x);
  u += 0x7fffu + ((u >> 16) & 1u);          // round-to-nearest-even
  return (unsigned short)(u >> 16);
}
__device__ __forceinline__ float bf2f(unsigned short h) {
  return __uint_as_float(((unsigned)h) << 16);
}
__device__ __forceinline__ f32x8 wmma_bf16(FragAB a, FragAB b, f32x8 c) {
  return __builtin_amdgcn_wmma_f32_16x16x32_bf16(false, a.v, false, b.v,
                                                 (short)0, c, false, false);
}

// ---------------------------------------------------------------------------
// bf16 GEMM:  C[M,N] = A[M,K] * Bt[N,K]^T   (Bt is B transposed, row-major)
// 128x128x32 tiles, 256 threads = 8 waves, each wave 64x32 (4x2 wmma tiles)
// ---------------------------------------------------------------------------
#define BM 128
#define BN 128
#define BK 32
#define LDT 40                                // padded LDS row stride (ushorts)

#define GF_BIAS 1
#define GF_ACC  2
#define GF_RELU 4
#define GF_F32  8
#define GF_BF16 16

__global__ __launch_bounds__(256)
void gemm_bf16_kernel(const unsigned short* __restrict__ A, int lda,
                      const unsigned short* __restrict__ Bt, int ldb,
                      const float* __restrict__ bias,
                      float* __restrict__ Cf, unsigned short* __restrict__ Cb,
                      int M, int N, int K, int flags)
{
  __shared__ unsigned short As[BM * LDT];
  __shared__ unsigned short Bs[BN * LDT];

  const int tid  = threadIdx.x;
  const int lane = tid & 31;
  const int wave = tid >> 5;                  // 0..7
  const int wm   = wave >> 2;                 // 0..1 along M
  const int wn   = wave & 3;                  // 0..3 along N
  const int bm   = blockIdx.x * BM;
  const int bn   = blockIdx.y * BN;
  const int kg   = lane >> 4;                 // half-wave group
  const int lr   = lane & 15;

  // global tile load mapping: 2 rows x 16 bytes per thread per tile
  const int lrow = tid >> 2;                  // 0..63
  const int lcol = (tid & 3) * 8;             // 0,8,16,24  (bf16 elems)

  f32x8 acc[4][2];
  const f32x8 zero = {};
#pragma unroll
  for (int i = 0; i < 4; ++i)
#pragma unroll
    for (int j = 0; j < 2; ++j) acc[i][j] = zero;

  const int KT = K / BK;

  f32x4 ra0, ra1, rb0, rb1;
  {
    const unsigned short* ap = A  + (size_t)(bm + lrow) * lda + lcol;
    const unsigned short* bp = Bt + (size_t)(bn + lrow) * ldb + lcol;
    ra0 = *(const f32x4*)ap;
    ra1 = *(const f32x4*)(ap + (size_t)64 * lda);
    rb0 = *(const f32x4*)bp;
    rb1 = *(const f32x4*)(bp + (size_t)64 * ldb);
  }
  *(f32x4*)&As[(size_t)lrow * LDT + lcol]        = ra0;
  *(f32x4*)&As[(size_t)(lrow + 64) * LDT + lcol] = ra1;
  *(f32x4*)&Bs[(size_t)lrow * LDT + lcol]        = rb0;
  *(f32x4*)&Bs[(size_t)(lrow + 64) * LDT + lcol] = rb1;
  __syncthreads();

  for (int kt = 0; kt < KT; ++kt) {
    if (kt + 1 < KT) {                         // next tile in flight during WMMA
      const unsigned short* ap = A  + (size_t)(bm + lrow) * lda + (kt + 1) * BK + lcol;
      const unsigned short* bp = Bt + (size_t)(bn + lrow) * ldb + (kt + 1) * BK + lcol;
      ra0 = *(const f32x4*)ap;
      ra1 = *(const f32x4*)(ap + (size_t)64 * lda);
      rb0 = *(const f32x4*)bp;
      rb1 = *(const f32x4*)(bp + (size_t)64 * ldb);
    }
    if (kt + 2 < KT) {                         // gfx1250 global_prefetch_b8
      __builtin_prefetch(A  + (size_t)(bm + lrow) * lda + (kt + 2) * BK + lcol, 0, 0);
      __builtin_prefetch(Bt + (size_t)(bn + lrow) * ldb + (kt + 2) * BK + lcol, 0, 0);
    }

    FragAB af[4], bfg[2];
#pragma unroll
    for (int i = 0; i < 4; ++i) {              // A 16x32 frag: lanes0-15 K{0..7,16..23}
      const unsigned short* p = &As[(size_t)(wm * 64 + i * 16 + lr) * LDT + kg * 8];
      af[i].q[0] = *(const f32x4*)p;
      af[i].q[1] = *(const f32x4*)(p + 16);
    }
#pragma unroll
    for (int j = 0; j < 2; ++j) {              // B 32x16 frag: lanes0-15 K{0..7,16..23}
      const unsigned short* p = &Bs[(size_t)(wn * 32 + j * 16 + lr) * LDT + kg * 8];
      bfg[j].q[0] = *(const f32x4*)p;
      bfg[j].q[1] = *(const f32x4*)(p + 16);
    }
#pragma unroll
    for (int i = 0; i < 4; ++i)
#pragma unroll
      for (int j = 0; j < 2; ++j) acc[i][j] = wmma_bf16(af[i], bfg[j], acc[i][j]);

    __syncthreads();
    if (kt + 1 < KT) {
      *(f32x4*)&As[(size_t)lrow * LDT + lcol]        = ra0;
      *(f32x4*)&As[(size_t)(lrow + 64) * LDT + lcol] = ra1;
      *(f32x4*)&Bs[(size_t)lrow * LDT + lcol]        = rb0;
      *(f32x4*)&Bs[(size_t)(lrow + 64) * LDT + lcol] = rb1;
      __syncthreads();
    }
  }

  // epilogue: C/D layout — vgpr p holds row kg*8+p, lanes0-15 = cols
#pragma unroll
  for (int i = 0; i < 4; ++i)
#pragma unroll
    for (int j = 0; j < 2; ++j) {
      const int col = bn + wn * 32 + j * 16 + lr;
      const float bv = (flags & GF_BIAS) ? bias[col] : 0.0f;
#pragma unroll
      for (int p = 0; p < 8; ++p) {
        const int row = bm + wm * 64 + i * 16 + kg * 8 + p;
        const size_t off = (size_t)row * N + col;
        float v = acc[i][j][p] + bv;
        if (flags & GF_ACC)  v += Cf[off];
        if (flags & GF_RELU) v  = fmaxf(v, 0.0f);
        if (flags & GF_F32)  Cf[off] = v;
        if (flags & GF_BF16) Cb[off] = f2bf(v);
      }
    }
}

// ---------------------------------------------------------------------------
// Criss-cross attention, one block per (a, b, head); 192 threads = 6 waves.
// Head h: qswap = rows are i (h1,h3) or j (h0,h2); keyfirst = K[b,k,a] vs
// K[b,a,k]; diag exclusion of key==row for h0,h1.  Softmax includes the
// prepended self score Q[b,i,j]·K[b,i,j].
// ---------------------------------------------------------------------------
#define ATTT 192

__global__ __launch_bounds__(ATTT)
void cc_attention_kernel(const unsigned short* __restrict__ Qb,
                         const unsigned short* __restrict__ Kb,
                         const unsigned short* __restrict__ Vb,
                         const float* __restrict__ mask,
                         unsigned short* __restrict__ Ctx)
{
  __shared__ unsigned short KV[DH_ * 104];     // K slab (96x192, stride 200) then Vt (192x96, stride 104)
  __shared__ unsigned short Ps[NS_ * 104];     // bf16 probabilities, stride 104
  __shared__ float mv[NS_];
  __shared__ float ss[NS_];
  __shared__ float ps[NS_];

  const int a  = blockIdx.x;
  const int bb = blockIdx.y;
  const int h  = blockIdx.z;
  const int qswap    = (h == 1) || (h == 3);
  const int keyfirst = (h == 1) || (h == 2);
  const int diag     = (h < 2);
  const float scale  = rsqrtf((float)DH_);

  const int tid  = threadIdx.x;
  const int lane = tid & 31;
  const int wave = tid >> 5;                   // row tile 0..5
  const int kg   = lane >> 4;
  const int lr   = lane & 15;

  const size_t baseBD = (size_t)bb * NS_ * NS_ * D_;
  auto selfOff = [&](int r) -> size_t {        // Q / self-K / self-V / ctx row
    const int i = qswap ? r : a, j = qswap ? a : r;
    return baseBD + ((size_t)i * NS_ + j) * D_ + (size_t)h * DH_;
  };
  auto keyOff = [&](int k) -> size_t {         // gathered key row
    const int x = keyfirst ? k : a, y = keyfirst ? a : k;
    return baseBD + ((size_t)x * NS_ + y) * D_ + (size_t)h * DH_;
  };

  // mask values + self scores
  if (tid < NS_) {
    const int k = tid;
    const int x = keyfirst ? k : a, y = keyfirst ? a : k;
    mv[k] = mask[((size_t)bb * NS_ + x) * NS_ + y];
    const size_t so = selfOff(k);
    const unsigned* qp = (const unsigned*)(Qb + so);
    const unsigned* kp = (const unsigned*)(Kb + so);
    float acc = 0.0f;
#pragma unroll 4
    for (int d = 0; d < DH_ / 2; ++d) {
      const unsigned qu = qp[d], ku = kp[d];
      acc += bf2f((unsigned short)qu)         * bf2f((unsigned short)ku);
      acc += bf2f((unsigned short)(qu >> 16)) * bf2f((unsigned short)(ku >> 16));
    }
    ss[k] = acc;
  }

  // stage K slab into LDS (Bt layout: key rows x DH, stride 200)
  for (int c = tid; c < NS_ * 24; c += ATTT) {
    const int krow = c / 24, cp = (c % 24) * 8;
    *(f32x4*)&KV[(size_t)krow * 200 + cp] = *(const f32x4*)(Kb + keyOff(krow) + cp);
  }
  __syncthreads();

  // S = Q K^T  (96x96) — per wave: 16 rows x 96 keys
  f32x8 Sacc[6];
  const f32x8 zero = {};
#pragma unroll
  for (int c = 0; c < 6; ++c) Sacc[c] = zero;

  for (int kd = 0; kd < 6; ++kd) {             // DH chunks of 32
    FragAB af;
    {
      const unsigned short* qp = Qb + selfOff(wave * 16 + lr) + kd * 32 + kg * 8;
      af.q[0] = *(const f32x4*)qp;
      af.q[1] = *(const f32x4*)(qp + 16);
    }
#pragma unroll
    for (int ct = 0; ct < 6; ++ct) {
      FragAB bfg;
      const unsigned short* p = &KV[(size_t)(ct * 16 + lr) * 200 + kd * 32 + kg * 8];
      bfg.q[0] = *(const f32x4*)p;
      bfg.q[1] = *(const f32x4*)(p + 16);
      Sacc[ct] = wmma_bf16(af, bfg, Sacc[ct]);
    }
  }

  // softmax over [self, 96 keys] — rows live in (kg, p), cols in lanes 0..15
#pragma unroll
  for (int p = 0; p < 8; ++p) {
    const int row = wave * 16 + kg * 8 + p;
    float vals[6];
    float m = -3.4e38f;
#pragma unroll
    for (int ct = 0; ct < 6; ++ct) {
      const int k = ct * 16 + lr;
      float me = mv[k];
      if (diag && k == row) me = 0.0f;
      const float v = Sacc[ct][p] * scale + (1.0f - me) * NEGV;
      vals[ct] = v;
      m = fmaxf(m, v);
    }
    for (int off = 1; off < 16; off <<= 1) m = fmaxf(m, __shfl_xor(m, off, 32));
    const float s0 = ss[row] * scale;
    m = fmaxf(m, s0);
    float sum = 0.0f;
#pragma unroll
    for (int ct = 0; ct < 6; ++ct) { vals[ct] = __expf(vals[ct] - m); sum += vals[ct]; }
    for (int off = 1; off < 16; off <<= 1) sum += __shfl_xor(sum, off, 32);
    const float e0 = __expf(s0 - m);
    sum += e0;
    const float inv = 1.0f / sum;
#pragma unroll
    for (int ct = 0; ct < 6; ++ct)
      Ps[(size_t)row * 104 + ct * 16 + lr] = f2bf(vals[ct] * inv);
    if (lr == 0) ps[row] = e0 * inv;
  }
  __syncthreads();

  // stage V transposed: Vt[d][k], stride 104 (reuses K LDS)
  for (int c = tid; c < NS_ * 24; c += ATTT) {
    const int krow = c / 24, cp = (c % 24) * 8;
    V4U8 t; t.v = *(const f32x4*)(Vb + keyOff(krow) + cp);
#pragma unroll
    for (int q = 0; q < 8; ++q) KV[(size_t)(cp + q) * 104 + krow] = t.u[q];
  }
  __syncthreads();

  // ctx = P V  (16 rows x 192)
  f32x8 Cc[12];
#pragma unroll
  for (int i = 0; i < 12; ++i) Cc[i] = zero;

  for (int kk = 0; kk < 3; ++kk) {             // key chunks of 32
    FragAB af;
    {
      const unsigned short* pp = &Ps[(size_t)(wave * 16 + lr) * 104 + kk * 32 + kg * 8];
      af.q[0] = *(const f32x4*)pp;
      af.q[1] = *(const f32x4*)(pp + 16);
    }
#pragma unroll
    for (int dt = 0; dt < 12; ++dt) {
      FragAB bfg;
      const unsigned short* p = &KV[(size_t)(dt * 16 + lr) * 104 + kk * 32 + kg * 8];
      bfg.q[0] = *(const f32x4*)p;
      bfg.q[1] = *(const f32x4*)(p + 16);
      Cc[dt] = wmma_bf16(af, bfg, Cc[dt]);
    }
  }

  // add self term, store ctx (bf16, head-concat layout)
#pragma unroll
  for (int p = 0; p < 8; ++p) {
    const int row = wave * 16 + kg * 8 + p;
    const size_t so = selfOff(row);
    const float pw = ps[row];
#pragma unroll
    for (int dt = 0; dt < 12; ++dt) {
      const int d = dt * 16 + lr;
      Ctx[so + d] = f2bf(Cc[dt][p] + pw * bf2f(Vb[so + d]));
    }
  }
}

// ---------------------------------------------------------------------------
// LayerNorm over last dim (768); optional residual add; fp32 and/or bf16 out
// ---------------------------------------------------------------------------
__global__ __launch_bounds__(256)
void ln_kernel(const float* __restrict__ X, const float* __restrict__ res,
               const float* __restrict__ g, const float* __restrict__ be,
               float* __restrict__ Of, unsigned short* __restrict__ Ob)
{
  __shared__ float red[8];
  const int row = blockIdx.x;
  const int tid = threadIdx.x;
  const size_t base = (size_t)row * D_;
  float v[3];
#pragma unroll
  for (int i = 0; i < 3; ++i) {
    const int c = tid + i * 256;
    v[i] = X[base + c];
    if (res) v[i] += res[base + c];
  }
  float s = v[0] + v[1] + v[2];
  for (int off = 1; off < 32; off <<= 1) s += __shfl_xor(s, off, 32);
  if ((tid & 31) == 0) red[tid >> 5] = s;
  __syncthreads();
  float tot = 0.0f;
#pragma unroll
  for (int w = 0; w < 8; ++w) tot += red[w];
  const float mean = tot / (float)D_;
  __syncthreads();
  float sq = 0.0f;
#pragma unroll
  for (int i = 0; i < 3; ++i) { const float d = v[i] - mean; sq += d * d; }
  for (int off = 1; off < 32; off <<= 1) sq += __shfl_xor(sq, off, 32);
  if ((tid & 31) == 0) red[tid >> 5] = sq;
  __syncthreads();
  float tv = 0.0f;
#pragma unroll
  for (int w = 0; w < 8; ++w) tv += red[w];
  const float rstd = rsqrtf(tv / (float)D_ + 1e-12f);
#pragma unroll
  for (int i = 0; i < 3; ++i) {
    const int c = tid + i * 256;
    const float o = (v[i] - mean) * rstd * g[c] + be[c];
    if (Of) Of[base + c] = o;
    if (Ob) Ob[base + c] = f2bf(o);
  }
}

// ---------------------------------------------------------------------------
// Elementwise kernels
// ---------------------------------------------------------------------------
__global__ void gate_kernel(const float* __restrict__ gs, const float* __restrict__ h2,
                            const float* __restrict__ i1, unsigned short* __restrict__ h3b,
                            size_t n)
{
  const size_t i = (size_t)blockIdx.x * 256 + threadIdx.x;
  if (i < n) {
    const float g = 1.0f / (1.0f + __expf(-gs[i]));
    h3b[i] = f2bf(g * h2[i] + (1.0f - g) * i1[i]);
  }
}

__global__ void f2bf_kernel(const float* __restrict__ x, unsigned short* __restrict__ y, size_t n)
{
  const size_t i = (size_t)blockIdx.x * 256 + threadIdx.x;
  if (i < n) y[i] = f2bf(x[i]);
}

// W (K x N) fp32  ->  Wt (N x K) bf16
__global__ void wtrans_kernel(const float* __restrict__ w, unsigned short* __restrict__ wt,
                              int K, int N)
{
  const size_t i = (size_t)blockIdx.x * 256 + threadIdx.x;
  if (i < (size_t)K * N) {
    const int k = (int)(i / N), nn = (int)(i % N);
    wt[(size_t)nn * K + k] = f2bf(w[i]);
  }
}

// ---------------------------------------------------------------------------
// Orchestration
// ---------------------------------------------------------------------------
extern "C" void kernel_launch(void* const* d_in, const int* in_sizes, int n_in,
                              void* d_out, int out_size, void* d_ws, size_t ws_size,
                              hipStream_t stream)
{
  (void)in_sizes; (void)n_in; (void)out_size; (void)ws_size;

  // Inputs: dict order for the top level; params assumed flattened in JAX
  // pytree order (dicts sorted by key).
  const float* inp  = (const float*)d_in[0];          // input
  const float* hs   = (const float*)d_in[1];          // entity_pair_matrix
  const float* mask = (const float*)d_in[2];          // masks
  const float* dn_b  = (const float*)d_in[3];
  const float* dn_be = (const float*)d_in[4];
  const float* dn_g  = (const float*)d_in[5];
  const float* dn_w  = (const float*)d_in[6];
  const float* c_b  = (const float*)d_in[7];
  const float* c_be = (const float*)d_in[8];
  const float* c_g  = (const float*)d_in[9];
  const float* c_w  = (const float*)d_in[10];
  const float* ca_kb = (const float*)d_in[11];
  const float* ca_kw = (const float*)d_in[12];
  const float* ca_ob = (const float*)d_in[13];
  const float* ca_ow = (const float*)d_in[14];
  const float* ca_qb = (const float*)d_in[15];
  const float* ca_qw = (const float*)d_in[16];
  const float* ca_vb = (const float*)d_in[17];
  const float* ca_vw = (const float*)d_in[18];
  const float* fh_b1 = (const float*)d_in[19];
  const float* fh_b2 = (const float*)d_in[20];
  const float* fh_be = (const float*)d_in[21];
  const float* fh_g  = (const float*)d_in[22];
  const float* fh_w1 = (const float*)d_in[23];
  const float* fh_w2 = (const float*)d_in[24];
  const float* fi_b1 = (const float*)d_in[25];
  const float* fi_b2 = (const float*)d_in[26];
  const float* fi_be = (const float*)d_in[27];
  const float* fi_g  = (const float*)d_in[28];
  const float* fi_w1 = (const float*)d_in[29];
  const float* fi_w2 = (const float*)d_in[30];
  const float* g_b  = (const float*)d_in[31];
  const float* g_w  = (const float*)d_in[32];
  const float* sa_kb = (const float*)d_in[33];
  const float* sa_kw = (const float*)d_in[34];
  const float* sa_ob = (const float*)d_in[35];
  const float* sa_ow = (const float*)d_in[36];
  const float* sa_qb = (const float*)d_in[37];
  const float* sa_qw = (const float*)d_in[38];
  const float* sa_vb = (const float*)d_in[39];
  const float* sa_vw = (const float*)d_in[40];

  // workspace layout
  char* wsp = (char*)d_ws;
  size_t off = 0;
  auto alloc = [&](size_t bytes) -> void* {
    void* p = wsp + off;
    off = (off + bytes + 255) & ~(size_t)255;
    return p;
  };
  const size_t MDf = (size_t)NTOK * D_ * sizeof(float);
  const size_t MDb = (size_t)NTOK * D_ * sizeof(unsigned short);
  const size_t MD2b = (size_t)NTOK * D2_ * sizeof(unsigned short);

  float* F0 = (float*)alloc(MDf);                 // gemm fp32 scratch
  float* F1 = (float*)alloc(MDf);                 // h2 fp32  (later aliased by x_bf)
  float* F2 = (float*)alloc(MDf);                 // i1 fp32  (later aliased by Hsn/Inpn bf16)
  unsigned short* B_inp = (unsigned short*)alloc(MDb);
  unsigned short* B_hs  = (unsigned short*)alloc(MDb);
  unsigned short* B_h1  = (unsigned short*)alloc(MDb);
  unsigned short* B_q   = (unsigned short*)alloc(MDb);
  unsigned short* B_k   = (unsigned short*)alloc(MDb);
  unsigned short* B_v   = (unsigned short*)alloc(MDb);
  unsigned short* B_ctx = (unsigned short*)alloc(MDb);
  unsigned short* B_h2  = (unsigned short*)alloc(MDb);
  unsigned short* B_i1  = (unsigned short*)alloc(MDb);
  // aliases (liveness-safe):
  unsigned short* B_h3   = B_h1;                  // h1 dead after cross K/V gemms
  unsigned short* B_x    = (unsigned short*)F1;   // M x 1536 bf16 == MDf bytes
  unsigned short* B_hsn  = (unsigned short*)F2;
  unsigned short* B_inpn = (unsigned short*)F2 + (size_t)NTOK * D_;
  (void)MD2b;

  auto wbuf = [&](size_t elems) { return (unsigned short*)alloc(elems * 2); };
  unsigned short* Wc  = wbuf((size_t)768 * 1536);
  unsigned short* Wqs = wbuf((size_t)768 * 768);
  unsigned short* Wks = wbuf((size_t)768 * 768);
  unsigned short* Wvs = wbuf((size_t)768 * 768);
  unsigned short* Wos = wbuf((size_t)768 * 768);
  unsigned short* Wqc = wbuf((size_t)768 * 768);
  unsigned short* Wkc = wbuf((size_t)768 * 768);
  unsigned short* Wvc = wbuf((size_t)768 * 768);
  unsigned short* Woc = wbuf((size_t)768 * 768);
  unsigned short* Wg  = wbuf((size_t)768 * 1536);
  unsigned short* W1h = wbuf((size_t)1536 * 768);
  unsigned short* W2h = wbuf((size_t)768 * 1536);
  unsigned short* W1i = wbuf((size_t)1536 * 768);
  unsigned short* W2i = wbuf((size_t)768 * 1536);
  unsigned short* Wd  = wbuf((size_t)768 * 1536);

  // launch helpers
  auto wtr = [&](const float* w, unsigned short* wt, int K, int N) {
    const size_t n = (size_t)K * N;
    wtrans_kernel<<<dim3((unsigned)((n + 255) / 256)), dim3(256), 0, stream>>>(w, wt, K, N);
  };
  auto cvt = [&](const float* x, unsigned short* y) {
    const size_t n = (size_t)NTOK * D_;
    f2bf_kernel<<<dim3((unsigned)((n + 255) / 256)), dim3(256), 0, stream>>>(x, y, n);
  };
  auto gemm = [&](const unsigned short* A, int lda, const unsigned short* Bt, int ldb,
                  const float* bias, float* Cf, unsigned short* Cb, int N, int K, int flags) {
    gemm_bf16_kernel<<<dim3(NTOK / BM, N / BN), dim3(256), 0, stream>>>(
        A, lda, Bt, ldb, bias, Cf, Cb, NTOK, N, K, flags);
  };
  auto ln = [&](const float* X, const float* res, const float* g, const float* be,
                float* Of, unsigned short* Ob) {
    ln_kernel<<<dim3(NTOK), dim3(256), 0, stream>>>(X, res, g, be, Of, Ob);
  };
  auto attn = [&](const unsigned short* Q, const unsigned short* K,
                  const unsigned short* V, unsigned short* C) {
    cc_attention_kernel<<<dim3(NS_, NB_, NH_), dim3(ATTT), 0, stream>>>(Q, K, V, mask, C);
  };

  // weights -> bf16 transposed
  wtr(c_w, Wc, 1536, 768);
  wtr(sa_qw, Wqs, 768, 768); wtr(sa_kw, Wks, 768, 768);
  wtr(sa_vw, Wvs, 768, 768); wtr(sa_ow, Wos, 768, 768);
  wtr(ca_qw, Wqc, 768, 768); wtr(ca_kw, Wkc, 768, 768);
  wtr(ca_vw, Wvc, 768, 768); wtr(ca_ow, Woc, 768, 768);
  wtr(g_w, Wg, 1536, 768);
  wtr(fh_w1, W1h, 768, 1536); wtr(fh_w2, W2h, 1536, 768);
  wtr(fi_w1, W1i, 768, 1536); wtr(fi_w2, W2i, 1536, 768);
  wtr(dn_w, Wd, 1536, 768);

  // activations -> bf16
  cvt(inp, B_inp);
  cvt(hs,  B_hs);

  // 1) h1 = LN(concat(Hs, Inp) @ Wc + bc)   (Hs first per reference)
  gemm(B_hs,  D_, Wc,        1536, nullptr, F0, nullptr, 768, 768, GF_F32);
  gemm(B_inp, D_, Wc + 768,  1536, c_b,     F0, nullptr, 768, 768, GF_ACC | GF_BIAS | GF_F32);
  ln(F0, nullptr, c_g, c_be, nullptr, B_h1);

  // 2) self attention: Q from Hs, K/V from h1
  gemm(B_hs, D_, Wqs, 768, sa_qb, nullptr, B_q, 768, 768, GF_BIAS | GF_BF16);
  gemm(B_h1, D_, Wks, 768, sa_kb, nullptr, B_k, 768, 768, GF_BIAS | GF_BF16);
  gemm(B_h1, D_, Wvs, 768, sa_vb, nullptr, B_v, 768, 768, GF_BIAS | GF_BF16);
  attn(B_q, B_k, B_v, B_ctx);
  gemm(B_ctx, D_, Wos, 768, sa_ob, F1, B_h2, 768, 768, GF_BIAS | GF_F32 | GF_BF16);

  // 3) cross attention: Q from Inp, K/V from h1
  gemm(B_inp, D_, Wqc, 768, ca_qb, nullptr, B_q, 768, 768, GF_BIAS | GF_BF16);
  gemm(B_h1,  D_, Wkc, 768, ca_kb, nullptr, B_k, 768, 768, GF_BIAS | GF_BF16);
  gemm(B_h1,  D_, Wvc, 768, ca_vb, nullptr, B_v, 768, 768, GF_BIAS | GF_BF16);
  attn(B_q, B_k, B_v, B_ctx);
  gemm(B_ctx, D_, Woc, 768, ca_ob, F2, B_i1, 768, 768, GF_BIAS | GF_F32 | GF_BF16);

  // 4) gate: g = sigmoid(concat(h2, i1) @ Wg + bg); h3 = g*h2 + (1-g)*i1
  gemm(B_h2, D_, Wg,       1536, nullptr, F0, nullptr, 768, 768, GF_F32);
  gemm(B_i1, D_, Wg + 768, 1536, g_b,     F0, nullptr, 768, 768, GF_ACC | GF_BIAS | GF_F32);
  {
    const size_t n = (size_t)NTOK * D_;
    gate_kernel<<<dim3((unsigned)((n + 255) / 256)), dim3(256), 0, stream>>>(F0, F1, F2, B_h3, n);
  }

  // 5) ffn_h -> Hs_new = LN(relu(h3 W1 + b1) W2 + b2 + Hs)
  gemm(B_h3, D_,  W1h, 768,  fh_b1, nullptr, B_x, 1536, 768,  GF_BIAS | GF_RELU | GF_BF16);
  gemm(B_x,  D2_, W2h, 1536, fh_b2, F0, nullptr, 768, 1536, GF_BIAS | GF_F32);
  ln(F0, hs, fh_g, fh_be, nullptr, B_hsn);

  // 6) ffn_i -> Inp_new
  gemm(B_h3, D_,  W1i, 768,  fi_b1, nullptr, B_x, 1536, 768,  GF_BIAS | GF_RELU | GF_BF16);
  gemm(B_x,  D2_, W2i, 1536, fi_b2, F0, nullptr, 768, 1536, GF_BIAS | GF_F32);
  ln(F0, inp, fi_g, fi_be, nullptr, B_inpn);

  // 7) dense: LN(concat(Inp_new, Hs_new) @ Wd + bd)  (Inp first per reference)
  gemm(B_inpn, D_, Wd,       1536, nullptr, F0, nullptr, 768, 768, GF_F32);
  gemm(B_hsn,  D_, Wd + 768, 1536, dn_b,    F0, nullptr, 768, 768, GF_ACC | GF_BIAS | GF_F32);
  ln(F0, nullptr, dn_g, dn_be, (float*)d_out, nullptr);
}
